// EnhancedBTIANet_67594195304461
// MI455X (gfx1250) — matrix operand order repaired and checked by reading, not compiled, not run on hardware
//
#include <hip/hip_runtime.h>
#include <hip/hip_bf16.h>
#include <math.h>

typedef __attribute__((ext_vector_type(16))) _Float16 v16h;
typedef __attribute__((ext_vector_type(8)))  _Float16 v8h;
typedef __attribute__((ext_vector_type(8)))  float    v8f;
typedef __attribute__((ext_vector_type(4)))  float    v4f;
typedef __attribute__((ext_vector_type(4)))  int      v4i;

static constexpr int Bv   = 1024;
static constexpr int Dv   = 768;
static constexpr int Hv   = 8;
static constexpr int Kv   = 10;
static constexpr int Nv   = 50000;
static constexpr int NvP  = 50048;          // Nv padded to a multiple of 64

#if defined(__has_builtin)
#if __has_builtin(__builtin_amdgcn_global_load_async_to_lds_b128) && \
    __has_builtin(__builtin_amdgcn_s_wait_asynccnt)
#define HAVE_ASYNC_LDS 1
#endif
#endif

__device__ __forceinline__ float gelu_f(float x) {
    return 0.5f * x * (1.0f + erff(x * 0.70710678118654752f));
}

__device__ __forceinline__ void stage_b128(const _Float16* gsrc, _Float16* ldst) {
#ifdef HAVE_ASYNC_LDS
    __builtin_amdgcn_global_load_async_to_lds_b128(
        (__attribute__((address_space(1))) v4i*)gsrc,
        (__attribute__((address_space(3))) v4i*)ldst, 0, 0);
#else
    *(v4f*)ldst = *(const v4f*)gsrc;   // 16B: global_load_b128 + ds_store_b128
#endif
}

__device__ __forceinline__ void wait_stage() {
#ifdef HAVE_ASYNC_LDS
    __builtin_amdgcn_s_wait_asynccnt(0);
#endif
}

// Load + convert the 16x32 f16 A fragment from f32 (ISA 7.12.2 16-bit A layout).
__device__ __forceinline__ v16h load_a_frag_f32(const float* arow, int k0) {
    const v4f* ap = (const v4f*)(arow + k0);
    v4f a0 = ap[0], a1 = ap[1], a2 = ap[4], a3 = ap[5];
    v16h af;
    #pragma unroll
    for (int i = 0; i < 4; ++i) {
        af[i]      = (_Float16)a0[i];
        af[4 + i]  = (_Float16)a1[i];
        af[8 + i]  = (_Float16)a2[i];
        af[12 + i] = (_Float16)a3[i];
    }
    return af;
}

// Contiguous-k A fragment (element s = arow[k0 + s]) from a pre-converted f16 A (two 16B loads, no converts).
__device__ __forceinline__ v16h load_a_frag_f16(const _Float16* arow, int k0) {
    v8h lo = *(const v8h*)(arow + k0);
    v8h hi = *(const v8h*)(arow + k0 + 8);
    v16h af;
    #pragma unroll
    for (int i = 0; i < 8; ++i) { af[i] = lo[i]; af[8 + i] = hi[i]; }
    return af;
}

// ---------------------------------------------------------------------------
// Generic WMMA GEMM (f32 weights, converted in-register), Nout % 64 == 0:
//   C[M,Nout] = epi(A[M,Kd] @ W[Nout,Kd]^T + bias (+resid))
// One 16x64 output tile per wave (4 accumulators, A fragment amortized 4x).
// EPI: 0 = +bias, 1 = gelu(+bias), 2 = +bias + resid.  OHALF: store f16.
// ---------------------------------------------------------------------------
template<int EPI, bool OHALF>
__global__ __launch_bounds__(256)
void gemm_wmma(const float* __restrict__ A, const float* __restrict__ W,
               const float* __restrict__ bias, const float* __restrict__ resid,
               void* __restrict__ Cv, int M, int Nout, int Kd)
{
    const int wave = threadIdx.x >> 5;
    const int lane = threadIdx.x & 31;
    const int ngrp = Nout >> 6;
    const long total = (long)(M >> 4) * ngrp;
    const long idx = (long)blockIdx.x * 8 + wave;
    if (idx >= total) return;                    // wave-uniform
    const int m0 = (int)(idx / ngrp) << 4;
    const int n0 = (int)(idx % ngrp) << 6;

    const int half = lane >> 4;
    const int l    = lane & 15;

    const float* arow = A + (long)(m0 + l) * Kd + (half ? 8 : 0);
    const float* wr[4];
    #pragma unroll
    for (int j = 0; j < 4; ++j)
        wr[j] = W + (long)(n0 + j * 16 + l) * Kd + (half ? 8 : 0);

    v8f acc[4] = {};
    for (int k0 = 0; k0 < Kd; k0 += 32) {
        v16h af = load_a_frag_f32(arow, k0);
        __builtin_prefetch(arow + k0 + 256, 0, 0);
        #pragma unroll
        for (int j = 0; j < 4; ++j) {
            const v4f* wp = (const v4f*)(wr[j] + k0);
            v4f w0 = wp[0], w1 = wp[1], w2 = wp[4], w3 = wp[5];
            v16h bf;
            #pragma unroll
            for (int i = 0; i < 4; ++i) {
                bf[i]      = (_Float16)w0[i];
                bf[4 + i]  = (_Float16)w1[i];
                bf[8 + i]  = (_Float16)w2[i];
                bf[12 + i] = (_Float16)w3[i];
            }
            acc[j] = __builtin_amdgcn_wmma_f32_16x16x32_f16(
                         false, af, false, bf, (short)0, acc[j], false, false);
        }
    }

    #pragma unroll
    for (int j = 0; j < 4; ++j) {
        const int col = n0 + j * 16 + l;
        const float bb = bias ? bias[col] : 0.0f;
        #pragma unroll
        for (int r = 0; r < 8; ++r) {
            const int row = m0 + r + (half ? 8 : 0);
            float v = acc[j][r] + bb;
            if constexpr (EPI == 1) v = gelu_f(v);
            if constexpr (EPI == 2) v += resid[(long)row * Nout + col];
            if constexpr (OHALF) ((_Float16*)Cv)[(long)row * Nout + col] = (_Float16)v;
            else                 ((float*)Cv)[(long)row * Nout + col] = v;
        }
    }
}

// ---------------------------------------------------------------------------
// Staged WMMA GEMM, f16 A and f16 W (the two Nout~50000 GEMMs).
// Block tile: 128 M-rows (8 waves x 16) x 64 N-cols. The shared 64x32-half
// B tile is staged into LDS double-buffered; each wave async-copies its 8 rows
// (32 lanes x 16B) per k-step, overlapping the next stage with compute.
// FULL: all 4 column tiles valid (no branches). gofs: column-group offset.
// W-row overreads past Nout stay inside the workspace (never consumed).
// ---------------------------------------------------------------------------
template<bool FULL>
__global__ __launch_bounds__(256)
void gemm_staged_h(const _Float16* __restrict__ A, const _Float16* __restrict__ W,
                   const float* __restrict__ bias, float* __restrict__ C,
                   int M, int Nout, int Kd, int gofs)
{
    __shared__ _Float16 lb[2][64 * 32];
    const int wave = threadIdx.x >> 5;
    const int lane = threadIdx.x & 31;
    const int half = lane >> 4;
    const int l    = lane & 15;
    const int hofs = half ? 16 : 0;

    const int ntn  = Nout >> 4;
    const int g    = blockIdx.x + gofs;
    const int n0   = g << 6;
    const int jmax = FULL ? 4 : min(4, ntn - (g << 2));
    const int m0   = blockIdx.y * 128 + wave * 16;

    const _Float16* arow = A + (long)(m0 + l) * Kd + (half ? 16 : 0);

    // Staging map: wave w fills rows [8w, 8w+8) of the 64-row B tile.
    const int srow   = wave * 8 + (lane >> 2);
    const int schunk = (lane & 3) * 8;                       // halves
    const _Float16* gsrc = W + (long)(n0 + srow) * Kd + schunk;
    _Float16* ld0 = &lb[0][srow * 32 + schunk];
    _Float16* ld1 = &lb[1][srow * 32 + schunk];

    v8f acc[4] = {};
    const int T = Kd / 32;
    stage_b128(gsrc, ld0);                                   // prologue: k-step 0
    for (int t = 0; t < T; ++t) {
        wait_stage();
        __syncthreads();                                     // buf[t&1] ready everywhere
        if (t + 1 < T)
            stage_b128(gsrc + (t + 1) * 32, (t & 1) ? ld0 : ld1);  // fill other buffer

        v16h af = load_a_frag_f16(arow, t * 32);
        const _Float16* lbase = lb[t & 1];
        if constexpr (FULL) {
            v16h b0 = *(const v16h*)&lbase[(0 + l) * 32 + hofs];
            v16h b1 = *(const v16h*)&lbase[(16 + l) * 32 + hofs];
            v16h b2 = *(const v16h*)&lbase[(32 + l) * 32 + hofs];
            v16h b3 = *(const v16h*)&lbase[(48 + l) * 32 + hofs];
            acc[0] = __builtin_amdgcn_wmma_f32_16x16x32_f16(false, af, false, b0, (short)0, acc[0], false, false);
            acc[1] = __builtin_amdgcn_wmma_f32_16x16x32_f16(false, af, false, b1, (short)0, acc[1], false, false);
            acc[2] = __builtin_amdgcn_wmma_f32_16x16x32_f16(false, af, false, b2, (short)0, acc[2], false, false);
            acc[3] = __builtin_amdgcn_wmma_f32_16x16x32_f16(false, af, false, b3, (short)0, acc[3], false, false);
        } else {
            #pragma unroll
            for (int j = 0; j < 4; ++j) {
                if (j < jmax) {
                    v16h bf = *(const v16h*)&lbase[(j * 16 + l) * 32 + hofs];
                    acc[j] = __builtin_amdgcn_wmma_f32_16x16x32_f16(
                                 false, af, false, bf, (short)0, acc[j], false, false);
                }
            }
        }
    }

    #pragma unroll
    for (int j = 0; j < 4; ++j) {
        if (FULL || j < jmax) {
            const int col = n0 + j * 16 + l;
            const float bb = bias ? bias[col] : 0.0f;
            #pragma unroll
            for (int r = 0; r < 8; ++r) {
                const int row = m0 + r + (half ? 8 : 0);
                C[(long)row * Nout + col] = acc[j][r] + bb;
            }
        }
    }
}

// ---------------------------------------------------------------------------
// LayerNorm: y = LN(x (+res)) * g + b, optional gelu. One block per row, D=768.
// ---------------------------------------------------------------------------
template<bool GELU>
__global__ __launch_bounds__(256)
void ln_kernel(const float* __restrict__ x, const float* __restrict__ res,
               const float* __restrict__ g, const float* __restrict__ b,
               float* __restrict__ y)
{
    const int row = blockIdx.x;
    const int t = threadIdx.x;
    const float* xr = x + (long)row * Dv;
    const float* rr = res ? res + (long)row * Dv : nullptr;
    __shared__ float red[12];

    float v[3];
    float s = 0.0f;
    #pragma unroll
    for (int i = 0; i < 3; ++i) {
        int d = t + i * 256;
        float val = xr[d] + (rr ? rr[d] : 0.0f);
        v[i] = val; s += val;
    }
    for (int off = 16; off > 0; off >>= 1) s += __shfl_xor(s, off, 32);
    if ((t & 31) == 0) red[t >> 5] = s;
    __syncthreads();
    if (t == 0) { float tot = 0; for (int i = 0; i < 8; ++i) tot += red[i]; red[8] = tot / Dv; }
    __syncthreads();
    const float mean = red[8];

    float vs = 0.0f;
    #pragma unroll
    for (int i = 0; i < 3; ++i) { float d0 = v[i] - mean; vs += d0 * d0; }
    for (int off = 16; off > 0; off >>= 1) vs += __shfl_xor(vs, off, 32);
    __syncthreads();
    if ((t & 31) == 0) red[t >> 5] = vs;
    __syncthreads();
    if (t == 0) { float tot = 0; for (int i = 0; i < 8; ++i) tot += red[i]; red[9] = tot / Dv; }
    __syncthreads();
    const float rstd = rsqrtf(red[9] + 1e-5f);

    float* yr = y + (long)row * Dv;
    #pragma unroll
    for (int i = 0; i < 3; ++i) {
        int d = t + i * 256;
        float o = (v[i] - mean) * rstd * g[d] + b[d];
        if constexpr (GELU) o = gelu_f(o);
        yr[d] = o;
    }
}

// ---------------------------------------------------------------------------
// Row L2-normalize (x / max(||x||, 1e-12)), output f32 or f16. Block per row.
// ---------------------------------------------------------------------------
template<bool TOHALF>
__global__ __launch_bounds__(256)
void rownorm_kernel(const float* __restrict__ x, void* __restrict__ y)
{
    const int row = blockIdx.x;
    const int t = threadIdx.x;
    const float* xr = x + (long)row * Dv;
    __shared__ float red[10];

    float v[3];
    float ss = 0.0f;
    #pragma unroll
    for (int i = 0; i < 3; ++i) { float val = xr[t + i * 256]; v[i] = val; ss += val * val; }
    for (int off = 16; off > 0; off >>= 1) ss += __shfl_xor(ss, off, 32);
    if ((t & 31) == 0) red[t >> 5] = ss;
    __syncthreads();
    if (t == 0) { float tot = 0; for (int i = 0; i < 8; ++i) tot += red[i]; red[8] = tot; }
    __syncthreads();
    const float sc = 1.0f / fmaxf(sqrtf(red[8]), 1e-12f);

    if constexpr (TOHALF) {
        _Float16* yr = (_Float16*)y + (long)row * Dv;
        #pragma unroll
        for (int i = 0; i < 3; ++i) yr[t + i * 256] = (_Float16)(v[i] * sc);
    } else {
        float* yr = (float*)y + (long)row * Dv;
        #pragma unroll
        for (int i = 0; i < 3; ++i) yr[t + i * 256] = v[i] * sc;
    }
}

__global__ __launch_bounds__(256)
void cvt_half_kernel(const float* __restrict__ x, _Float16* __restrict__ y, long n)
{
    long i = (long)blockIdx.x * 256 + threadIdx.x;
    long stride = (long)gridDim.x * 256;
    for (; i < n; i += stride) y[i] = (_Float16)x[i];
}

__global__ __launch_bounds__(256)
void concat_kernel(const float* __restrict__ a, const float* __restrict__ b,
                   float* __restrict__ y)
{
    long i = (long)blockIdx.x * 256 + threadIdx.x;
    long n = (long)Bv * 2 * Dv;
    long stride = (long)gridDim.x * 256;
    for (; i < n; i += stride) {
        long row = i / (2 * Dv);
        int  col = (int)(i % (2 * Dv));
        y[i] = (col < Dv) ? a[row * Dv + col] : b[row * Dv + (col - Dv)];
    }
}

__device__ __forceinline__ void topk_insert(float v, int id, float* lv, int* li)
{
    if (v <= lv[Kv - 1]) return;
    int p = Kv - 1;
    while (p > 0 && v > lv[p - 1]) { lv[p] = lv[p - 1]; li[p] = li[p - 1]; --p; }
    lv[p] = v; li[p] = id;
}

__global__ __launch_bounds__(256)
void topk_kernel(const float* __restrict__ sim, int* __restrict__ outidx, int stride)
{
    const int row = blockIdx.x;
    const float* sr = sim + (long)row * stride;
    float lv[Kv]; int li[Kv];
    #pragma unroll
    for (int i = 0; i < Kv; ++i) { lv[i] = -3.4e38f; li[i] = 0; }
    for (int n = threadIdx.x; n < Nv; n += 256) topk_insert(sr[n], n, lv, li);

    __shared__ float sv[256 * Kv];
    __shared__ int   si[256 * Kv];
    #pragma unroll
    for (int i = 0; i < Kv; ++i) { sv[threadIdx.x * Kv + i] = lv[i]; si[threadIdx.x * Kv + i] = li[i]; }
    __syncthreads();
    if (threadIdx.x == 0) {
        float bv[Kv]; int bi[Kv];
        #pragma unroll
        for (int i = 0; i < Kv; ++i) { bv[i] = -3.4e38f; bi[i] = 0; }
        for (int j = 0; j < 256 * Kv; ++j) topk_insert(sv[j], si[j], bv, bi);
        for (int i = 0; i < Kv; ++i) outidx[row * Kv + i] = bi[i];
    }
}

__global__ __launch_bounds__(256)
void gather_kernel(const float* __restrict__ emb, const int* __restrict__ idx,
                   float* __restrict__ out)
{
    const long r = blockIdx.x;                 // 0 .. B*K-1
    const long src = idx[r];
    for (int d = threadIdx.x; d < Dv; d += 256)
        out[r * Dv + d] = emb[src * Dv + d];
}

// MHA #4 attention core: block per batch, wave per head, dh = 96 (3 per lane).
__global__ __launch_bounds__(256)
void attn_kernel(const float* __restrict__ qh, const float* __restrict__ kh,
                 const float* __restrict__ vh, float* __restrict__ o)
{
    const int b = blockIdx.x, w = threadIdx.x >> 5, lane = threadIdx.x & 31;
    const int dh = Dv / Hv;                    // 96
    const float scale = 0.1020620726159658f;   // 1/sqrt(96)
    const float* q = qh + (long)b * Dv + w * dh + lane * 3;
    float q0 = q[0], q1 = q[1], q2 = q[2];

    float sc[Kv];
    #pragma unroll
    for (int k = 0; k < Kv; ++k) {
        const float* kp = kh + ((long)b * Kv + k) * Dv + w * dh + lane * 3;
        float p = q0 * kp[0] + q1 * kp[1] + q2 * kp[2];
        for (int off = 16; off > 0; off >>= 1) p += __shfl_xor(p, off, 32);
        sc[k] = p * scale;
    }
    float m = sc[0];
    #pragma unroll
    for (int k = 1; k < Kv; ++k) m = fmaxf(m, sc[k]);
    float sum = 0.0f;
    #pragma unroll
    for (int k = 0; k < Kv; ++k) { sc[k] = expf(sc[k] - m); sum += sc[k]; }
    const float inv = 1.0f / sum;

    float o0 = 0, o1 = 0, o2 = 0;
    #pragma unroll
    for (int k = 0; k < Kv; ++k) {
        const float* vp = vh + ((long)b * Kv + k) * Dv + w * dh + lane * 3;
        float a = sc[k] * inv;
        o0 += a * vp[0]; o1 += a * vp[1]; o2 += a * vp[2];
    }
    float* op = o + (long)b * Dv + w * dh + lane * 3;
    op[0] = o0; op[1] = o1; op[2] = o2;
}

// ---------------------------------------------------------------------------
static void launch_gemm(int epi, bool ohalf, const float* A, const float* W,
                        const float* bias, const float* resid, void* C,
                        int M, int Nn, int Kd, hipStream_t s)
{
    long total = (long)(M / 16) * (Nn / 64);
    dim3 grid((unsigned)((total + 7) / 8)), block(256);
    if (ohalf)         gemm_wmma<1, true ><<<grid, block, 0, s>>>(A, W, bias, resid, C, M, Nn, Kd);
    else if (epi == 0) gemm_wmma<0, false><<<grid, block, 0, s>>>(A, W, bias, resid, C, M, Nn, Kd);
    else if (epi == 1) gemm_wmma<1, false><<<grid, block, 0, s>>>(A, W, bias, resid, C, M, Nn, Kd);
    else               gemm_wmma<2, false><<<grid, block, 0, s>>>(A, W, bias, resid, C, M, Nn, Kd);
}

static void launch_gemm_h(const _Float16* A, const _Float16* W, const float* bias,
                          float* C, int M, int Nn, int Kd, hipStream_t s)
{
    int ntn = Nn / 16, full = ntn / 4;
    dim3 block(256);
    if (full > 0)
        gemm_staged_h<true ><<<dim3(full, M / 128), block, 0, s>>>(A, W, bias, C, M, Nn, Kd, 0);
    if (ntn % 4)
        gemm_staged_h<false><<<dim3(1, M / 128), block, 0, s>>>(A, W, bias, C, M, Nn, Kd, full);
}

extern "C" void kernel_launch(void* const* d_in, const int* in_sizes, int n_in,
                              void* d_out, int out_size, void* d_ws, size_t ws_size,
                              hipStream_t stream)
{
    const float* visual   = (const float*)d_in[0];
    const float* text     = (const float*)d_in[1];
    const float* ans_emb  = (const float*)d_in[2];
    const float* vqa_in_w = (const float*)d_in[3];
    const float* vqa_in_b = (const float*)d_in[4];
    const float* vqa_ow   = (const float*)d_in[5];
    const float* vqa_ob   = (const float*)d_in[6];
    const float* fproj_w  = (const float*)d_in[7];
    const float* fproj_b  = (const float*)d_in[8];
    const float* fln_g    = (const float*)d_in[9];
    const float* fln_b    = (const float*)d_in[10];
    const float* sim_w    = (const float*)d_in[11];
    const float* sim_b    = (const float*)d_in[12];
    const float* mha_iw   = (const float*)d_in[13];
    const float* mha_ib   = (const float*)d_in[14];
    const float* mha_ow   = (const float*)d_in[15];
    const float* mha_ob   = (const float*)d_in[16];
    const float* ffn_w1   = (const float*)d_in[17];
    const float* ffn_b1   = (const float*)d_in[18];
    const float* ffn_w2   = (const float*)d_in[19];
    const float* ffn_b2   = (const float*)d_in[20];
    const float* ln_g     = (const float*)d_in[21];
    const float* ln_b     = (const float*)d_in[22];
    const float* outp_w   = (const float*)d_in[23];
    const float* outp_b   = (const float*)d_in[24];
    const float* open_w1  = (const float*)d_in[25];
    const float* open_b1  = (const float*)d_in[26];
    const float* open_w2  = (const float*)d_in[27];
    const float* open_b2  = (const float*)d_in[28];
    float* out = (float*)d_out;

    const size_t BD  = (size_t)Bv * Dv;
    const size_t ND  = (size_t)Nv * Dv;
    const size_t NDP = (size_t)NvP * Dv;
    float* f = (float*)d_ws;
    float* t0    = f + 0 * BD;
    float* attq  = f + 1 * BD;
    float* cat   = f + 2 * BD;          // 2*BD
    float* fpre  = f + 4 * BD;
    float* fused = f + 5 * BD;
    float* fproj = f + 6 * BD;
    _Float16* fn16 = (_Float16*)(f + 7 * BD);   // B x D halves
    float* v1    = f + 8 * BD;
    float* t1    = f + 9 * BD;
    float* a2    = f + 10 * BD;
    float* a3    = f + 11 * BD;
    float* fz0   = f + 12 * BD;
    float* qh    = f + 13 * BD;
    float* atto  = f + 14 * BD;
    float* ag    = f + 15 * BD;
    float* fz1   = f + 16 * BD;
    float* outv  = f + 17 * BD;
    _Float16* h16 = (_Float16*)(f + 18 * BD);   // B x D halves
    float* fz2   = f + 19 * BD;
    float* tmpu  = f + 20 * BD;
    float* mid   = f + 21 * BD;         // B x 4D -> 4*BD
    float* tke   = f + 25 * BD;         // (B*K) x D -> 10*BD
    float* khb   = f + 35 * BD;         // 10*BD
    float* vhb   = f + 45 * BD;         // 10*BD
    int*   idxb  = (int*)(f + 55 * BD); // B*K ints
    _Float16* an16 = (_Float16*)(f + 56 * BD);              // NvP x D halves
    _Float16* w216 = (_Float16*)(f + 56 * BD + NDP / 2);    // Nv x D halves
    float* sim   = f + 56 * BD + NDP / 2 + ND / 2;          // B x NvP floats

    // --- seq-1 MHA (vqa): attn_q = (visual @ Wv^T + bv) @ out_w^T + out_b ---
    launch_gemm(0, false, visual, vqa_in_w + (size_t)2 * Dv * Dv, vqa_in_b + 2 * Dv,
                nullptr, t0, Bv, Dv, Dv, stream);
    launch_gemm(0, false, t0, vqa_ow, vqa_ob, nullptr, attq, Bv, Dv, Dv, stream);

    // --- fused = gelu(LN(concat(visual, attn_q) @ fproj_w^T + fproj_b)) ---
    concat_kernel<<<2048, 256, 0, stream>>>(visual, attq, cat);
    launch_gemm(0, false, cat, fproj_w, fproj_b, nullptr, fpre, Bv, Dv, 2 * Dv, stream);
    ln_kernel<true><<<Bv, 256, 0, stream>>>(fpre, nullptr, fln_g, fln_b, fused);

    // --- sim = normalize(fused @ sim_w^T + sim_b) @ normalize(ans_emb)^T ---
    launch_gemm(0, false, fused, sim_w, sim_b, nullptr, fproj, Bv, Dv, Dv, stream);
    rownorm_kernel<true><<<Bv, 256, 0, stream>>>(fproj, fn16);
    rownorm_kernel<true><<<Nv, 256, 0, stream>>>(ans_emb, an16);
    launch_gemm_h(fn16, an16, nullptr, sim, Bv, NvP, Dv, stream);   // padded N, no tail

    // --- top-K indices, gather answer embeddings ---
    topk_kernel<<<Bv, 256, 0, stream>>>(sim, idxb, NvP);
    gather_kernel<<<Bv * Kv, 256, 0, stream>>>(ans_emb, idxb, tke);

    // --- seq-1 MHAs 0..3 + LayerNorms ---
    const size_t IW = (size_t)3 * Dv * Dv;  // per-layer mha_in_w stride
    const size_t OW = (size_t)Dv * Dv;
    // m0 on visual -> v1
    launch_gemm(0, false, visual, mha_iw + 0 * IW + 2 * OW, mha_ib + 0 * 3 * Dv + 2 * Dv,
                nullptr, tmpu, Bv, Dv, Dv, stream);
    launch_gemm(0, false, tmpu, mha_ow + 0 * OW, mha_ob + 0 * Dv, nullptr, t0, Bv, Dv, Dv, stream);
    ln_kernel<false><<<Bv, 256, 0, stream>>>(t0, visual, ln_g + 0 * Dv, ln_b + 0 * Dv, v1);
    // m1 on text -> t1
    launch_gemm(0, false, text, mha_iw + 1 * IW + 2 * OW, mha_ib + 1 * 3 * Dv + 2 * Dv,
                nullptr, tmpu, Bv, Dv, Dv, stream);
    launch_gemm(0, false, tmpu, mha_ow + 1 * OW, mha_ob + 1 * Dv, nullptr, t0, Bv, Dv, Dv, stream);
    ln_kernel<false><<<Bv, 256, 0, stream>>>(t0, text, ln_g + 1 * Dv, ln_b + 1 * Dv, t1);
    // m2(v1, t1, t1) -> a2 (only V path matters, seq-1)
    launch_gemm(0, false, t1, mha_iw + 2 * IW + 2 * OW, mha_ib + 2 * 3 * Dv + 2 * Dv,
                nullptr, tmpu, Bv, Dv, Dv, stream);
    launch_gemm(0, false, tmpu, mha_ow + 2 * OW, mha_ob + 2 * Dv, nullptr, a2, Bv, Dv, Dv, stream);
    // m3(t1, v1, v1) -> a3
    launch_gemm(0, false, v1, mha_iw + 3 * IW + 2 * OW, mha_ib + 3 * 3 * Dv + 2 * Dv,
                nullptr, tmpu, Bv, Dv, Dv, stream);
    launch_gemm(0, false, tmpu, mha_ow + 3 * OW, mha_ob + 3 * Dv, nullptr, a3, Bv, Dv, Dv, stream);
    ln_kernel<false><<<Bv, 256, 0, stream>>>(a2, a3, ln_g + 2 * Dv, ln_b + 2 * Dv, fz0);

    // --- MHA 4 (real attention over K=10 gathered keys) ---
    launch_gemm(0, false, fz0, mha_iw + 4 * IW + 0 * OW, mha_ib + 4 * 3 * Dv + 0 * Dv,
                nullptr, qh, Bv, Dv, Dv, stream);
    launch_gemm(0, false, tke, mha_iw + 4 * IW + 1 * OW, mha_ib + 4 * 3 * Dv + 1 * Dv,
                nullptr, khb, Bv * Kv, Dv, Dv, stream);
    launch_gemm(0, false, tke, mha_iw + 4 * IW + 2 * OW, mha_ib + 4 * 3 * Dv + 2 * Dv,
                nullptr, vhb, Bv * Kv, Dv, Dv, stream);
    attn_kernel<<<Bv, 256, 0, stream>>>(qh, khb, vhb, atto);
    launch_gemm(0, false, atto, mha_ow + 4 * OW, mha_ob + 4 * Dv, nullptr, ag, Bv, Dv, Dv, stream);
    ln_kernel<false><<<Bv, 256, 0, stream>>>(ag, fz0, ln_g + 3 * Dv, ln_b + 3 * Dv, fz1);

    // --- FFN with residual ---
    launch_gemm(1, false, fz1, ffn_w1, ffn_b1, nullptr, mid, Bv, 4 * Dv, Dv, stream);
    launch_gemm(2, false, mid, ffn_w2, ffn_b2, fz1, fz2, Bv, Dv, 4 * Dv, stream);

    // --- heads ---
    launch_gemm(0, false, fz2, outp_w, outp_b, nullptr, outv, Bv, Dv, Dv, stream);
    launch_gemm(1, true, outv, open_w1, open_b1, nullptr, h16, Bv, Dv, Dv, stream); // f16 out
    cvt_half_kernel<<<4096, 256, 0, stream>>>(open_w2, w216, (long)ND);
    launch_gemm_h(h16, w216, open_b2, out, Bv, Nv, Dv, stream);     // 781 full + 1 tail
}
